// ResidualVectorQuantizer_2405181686101
// MI455X (gfx1250) — matrix operand order repaired and checked by reading, not compiled, not run on hardware
//
#include <hip/hip_runtime.h>
#include <hip/hip_bf16.h>

// ---------------------------------------------------------------------------
// Residual Vector Quantizer forward for MI455X (gfx1250, wave32, WMMA + TDM).
//
// Roofline: ~620 GFLOP total, dominated by 3x [32768x512x4096] codebook-score
// GEMMs; memory traffic ~0.5 GB (=~20us @ 23.3 TB/s) => matrix-core bound.
// => everything runs on v_wmma_f32_16x16x32_bf16 (f32 accum), weights and
// codebooks pre-converted to bf16, B tiles streamed into LDS by the Tensor
// Data Mover (tensor_load_to_lds / s_wait_tensorcnt), double-buffered.
// ---------------------------------------------------------------------------

typedef __attribute__((ext_vector_type(16))) __bf16 v16bf;
typedef __attribute__((ext_vector_type(8)))  __bf16 v8bf;
typedef __attribute__((ext_vector_type(8)))  float  v8f;
typedef __attribute__((ext_vector_type(4)))  unsigned int u32x4;
typedef __attribute__((ext_vector_type(8)))  int    i32x8;
typedef __attribute__((ext_vector_type(4)))  int    i32x4;

#define N_TOK 32768
#define D_IN  1024
#define HDIM  512
#define KCODE 4096
#define NCB   3

// ---------------------------------------------------------------------------
// Tensor Data Mover: DMA a 128x32 bf16 tile of a 2D row-major tensor into LDS.
// D# group0: count | lds_addr | global_addr | type=2
// D# group1: data_size=2B, tensor_dim0/1, tile_dim0=32, tile_dim1=128, stride.
// ---------------------------------------------------------------------------
#if __has_builtin(__builtin_amdgcn_tensor_load_to_lds)
#define HAVE_TDM 1
#if __has_builtin(__builtin_amdgcn_s_wait_tensorcnt)
#define TDM_WAIT(n) __builtin_amdgcn_s_wait_tensorcnt(n)
#else
#define TDM_WAIT(n) asm volatile("s_wait_tensorcnt %0" ::"i"(n))
#endif

static __device__ __forceinline__ void tdm_load_tile_128x32(
    const __bf16* gbase, long row0, long col0, int ld /*elems*/, int nrows,
    unsigned lds_byte_off) {
  unsigned long long ga = (unsigned long long)(gbase + row0 * (long)ld + col0);
  u32x4 g0;
  g0.x = 1u;                                             // count=1 (user D#)
  g0.y = lds_byte_off;                                   // lds_addr
  g0.z = (unsigned)ga;                                   // global_addr[31:0]
  g0.w = (unsigned)((ga >> 32) & 0x01FFFFFFull) | (2u << 30);  // addr hi | type=2
  i32x8 g1;
  g1[0] = (int)(1u << 16);                               // data_size=1 -> 2 bytes
  g1[1] = (int)(((unsigned)ld & 0xFFFFu) << 16);         // tensor_dim0[15:0]
  g1[2] = (int)((((unsigned)ld >> 16) & 0xFFFFu) |
                (((unsigned)nrows & 0xFFFFu) << 16));    // dim0 hi | dim1 lo
  g1[3] = (int)((((unsigned)nrows >> 16) & 0xFFFFu) |
                (32u << 16));                            // dim1 hi | tile_dim0=32
  g1[4] = 128;                                           // tile_dim1=128
  g1[5] = ld;                                            // tensor_dim0_stride lo32
  g1[6] = 0;
  g1[7] = 0;
  i32x4 z4 = (i32x4){0, 0, 0, 0};
#if defined(__clang_major__) && (__clang_major__ >= 23)
  i32x8 z8 = (i32x8){0, 0, 0, 0, 0, 0, 0, 0};
  __builtin_amdgcn_tensor_load_to_lds(g0, g1, z4, z4, z8, 0);   // 6-arg toolchain
#else
  __builtin_amdgcn_tensor_load_to_lds(g0, g1, z4, z4, 0);       // 5-arg ROCm 7.2
#endif
}
#else
#define HAVE_TDM 0
#define TDM_WAIT(n)
#endif

// Map float to a uint whose unsigned order matches float order.
static __device__ __forceinline__ unsigned order_f32(float f) {
  unsigned u = __float_as_uint(f);
  return (u & 0x80000000u) ? ~u : (u | 0x80000000u);
}

// Build a 16-bit A fragment in contiguous k order: lane<16 holds K{0..15},
// lane>=16 holds K{16..31}: two 16B chunks at +0 / +16 bytes.
static __device__ __forceinline__ v16bf load_a_frag(const __bf16* row_k0,
                                                    int lhalf) {
  const __bf16* p = row_k0 + lhalf * 16;
  v8bf lo = *(const v8bf*)p;
  v8bf hi = *(const v8bf*)(p + 8);
  v16bf r;
#pragma unroll
  for (int e = 0; e < 8; ++e) { r[e] = lo[e]; r[8 + e] = hi[e]; }
  return r;
}

// ---------------------------------------------------------------------------
// f32 -> bf16 plain convert (codebooks)
// ---------------------------------------------------------------------------
__global__ __launch_bounds__(256) void cvt_f32_bf16(const float* __restrict__ src,
                                                    __bf16* __restrict__ dst, int n) {
  int i = blockIdx.x * 256 + threadIdx.x;
  if (i < n) dst[i] = (__bf16)src[i];
}

// f32 [Kd][Nd] -> bf16 transposed [Nd][Kd]  (weights, once per launch)
__global__ __launch_bounds__(256) void cvt_transpose_kernel(
    const float* __restrict__ src, __bf16* __restrict__ dst, int Kd, int Nd) {
  long i = (long)blockIdx.x * 256 + threadIdx.x;
  if (i < (long)Kd * Nd) {
    long n = i / Kd, k = i - n * Kd;
    dst[i] = (__bf16)src[k * (long)Nd + n];
  }
}

// Per-code squared norms
__global__ __launch_bounds__(256) void cb_norm_kernel(const float* __restrict__ cb,
                                                      float* __restrict__ norms,
                                                      int rows, int hdim) {
  int r = blockIdx.x * 256 + threadIdx.x;
  if (r >= rows) return;
  const float* p = cb + (long)r * hdim;
  float s = 0.f;
  for (int k = 0; k < hdim; ++k) { float v = p[k]; s += v * v; }
  norms[r] = s;
}

// ---------------------------------------------------------------------------
// Fused GEMM: C[M,Nd] = act(A[M,Kd] @ W + bias); W pre-transposed bf16 [Nd][Kd].
// 128x128 block, 8 waves (4Mx2N), wave 32x64 = 2x4 grid of 16x16x32 WMMA.
// Double-buffered LDS; B tiles via TDM one K-step ahead.
// ---------------------------------------------------------------------------
template <bool RELU>
__global__ __launch_bounds__(256) void gemm_bias_kernel(
    const float* __restrict__ A, const __bf16* __restrict__ Wt,
    const float* __restrict__ bias, float* __restrict__ C, int Kd, int Nd) {
  __shared__ __bf16 aT[2][128][32];
  __shared__ __bf16 bT[2][128][32];

  const int tid   = threadIdx.x;
  const int lane  = tid & 31;
  const int wv    = tid >> 5;
  const int wm    = wv & 3;
  const int wn    = wv >> 2;
  const int lhalf = lane >> 4;
  const int l15   = lane & 15;
  const long blockM = (long)blockIdx.y * 128;
  const long blockN = (long)blockIdx.x * 128;
  const int  T      = Kd >> 5;

  v8f acc[2][4];
  for (int mi = 0; mi < 2; ++mi)
    for (int nj = 0; nj < 4; ++nj)
      for (int e = 0; e < 8; ++e) acc[mi][nj][e] = 0.f;

  auto stageA = [&](int t, int buf) {
#pragma unroll
    for (int i = 0; i < 16; ++i) {
      int e = tid + i * 256;
      int m = e >> 5, k = e & 31;
      aT[buf][m][k] = (__bf16)A[(blockM + m) * (long)Kd + t * 32 + k];
    }
  };
  auto stageB = [&](int t, int buf) {
#if HAVE_TDM
    if (tid < 32)   // one wave issues the DMA descriptor
      tdm_load_tile_128x32(Wt, blockN, (long)t * 32, Kd, Nd,
                           (unsigned)(size_t)&bT[buf][0][0]);
#else
#pragma unroll
    for (int i = 0; i < 16; ++i) {
      int e = tid + i * 256;
      int n = e >> 5, k = e & 31;
      bT[buf][n][k] = Wt[(blockN + n) * (long)Kd + t * 32 + k];
    }
#endif
  };

  stageA(0, 0);
  stageB(0, 0);
  for (int t = 0; t < T; ++t) {
    const int cur = t & 1, nxt = cur ^ 1;
    if (t + 2 < T)
      __builtin_prefetch(&A[(blockM + wv * 16) * (long)Kd + (t + 2) * 32], 0, 0);
    if (t + 1 < T) { stageA(t + 1, nxt); stageB(t + 1, nxt); }
#if HAVE_TDM
    if (tid < 32) {
      if (t + 1 < T) TDM_WAIT(1);   // tile t landed; t+1 may be in flight
      else           TDM_WAIT(0);
    }
#endif
    __syncthreads();                 // tile t visible to everyone

    v16bf af[2], bv[4];
#pragma unroll
    for (int mi = 0; mi < 2; ++mi)
      af[mi] = load_a_frag(&aT[cur][wm * 32 + mi * 16 + l15][0], lhalf);
#pragma unroll
    for (int nj = 0; nj < 4; ++nj)
      bv[nj] = *(const v16bf*)&bT[cur][wn * 64 + nj * 16 + l15][lhalf * 16];

#pragma unroll
    for (int mi = 0; mi < 2; ++mi)
#pragma unroll
      for (int nj = 0; nj < 4; ++nj)
        acc[mi][nj] = __builtin_amdgcn_wmma_f32_16x16x32_bf16(
            false, af[mi], false, bv[nj], (short)0, acc[mi][nj], false, false);

    __syncthreads();                 // readers done before buf is restaged
  }

#pragma unroll
  for (int mi = 0; mi < 2; ++mi)
#pragma unroll
    for (int nj = 0; nj < 4; ++nj) {
      long col = blockN + wn * 64 + nj * 16 + l15;
      float bvs = bias[col];
#pragma unroll
      for (int r = 0; r < 8; ++r) {
        long row = blockM + wm * 32 + mi * 16 + lhalf * 8 + r;
        float v = acc[mi][nj][r] + bvs;
        if (RELU) v = v > 0.f ? v : 0.f;
        C[row * (long)Nd + col] = v;
      }
    }
}

// ---------------------------------------------------------------------------
// One RVQ stage. Residual tile (128x512) converted to bf16 ONCE into LDS
// (128 KB of the 320 KB WGP LDS); codebook tiles streamed by TDM. Score GEMM
// in bf16 WMMA, argmin via packed-key ds_min_u64, then gather + update.
// Dynamic LDS: 128*512*2 + 2*128*32*2 + 128*8 = 148480 B.
// ---------------------------------------------------------------------------
__global__ __launch_bounds__(256) void rvq_stage_kernel(
    float* __restrict__ residual, float* __restrict__ quant,
    float* __restrict__ idx_out, const __bf16* __restrict__ cbb,
    const float* __restrict__ cbn, const float* __restrict__ cbf, int first) {
  extern __shared__ char smem[];
  __bf16* rb = (__bf16*)smem;                                     // [128][512]
  __bf16(*bT)[128][32] = (__bf16(*)[128][32])(smem + 128 * 512 * 2);
  unsigned long long* best =
      (unsigned long long*)(smem + 128 * 512 * 2 + 2 * 128 * 32 * 2);

  const int tid   = threadIdx.x;
  const int lane  = tid & 31;
  const int wv    = tid >> 5;
  const int wm    = wv & 3;
  const int wn    = wv >> 2;
  const int lhalf = lane >> 4;
  const int l15   = lane & 15;
  const long tok0 = (long)blockIdx.x * 128;

  // Stage residual tile once: f32 -> bf16, 65536 elems / 256 threads.
  for (int i = 0; i < 256; ++i) {
    int e = tid + i * 256;
    int m = e >> 9, k = e & 511;
    rb[m * 512 + k] = (__bf16)residual[(tok0 + m) * (long)HDIM + k];
  }
  if (tid < 128) best[tid] = 0xFFFFFFFFFFFFFFFFull;
  __syncthreads();

  auto stageB = [&](int nt, int t, int buf) {
#if HAVE_TDM
    if (tid < 32)
      tdm_load_tile_128x32(cbb, (long)nt * 128, (long)t * 32, HDIM, KCODE,
                           (unsigned)(size_t)&bT[buf][0][0]);
#else
#pragma unroll
    for (int i = 0; i < 16; ++i) {
      int e = tid + i * 256;
      int n = e >> 5, k = e & 31;
      bT[buf][n][k] = cbb[(long)(nt * 128 + n) * HDIM + t * 32 + k];
    }
#endif
  };

  const int T = HDIM / 32;  // 16 K-steps
  for (int nt = 0; nt < KCODE / 128; ++nt) {
    v8f acc[2][4];
    for (int mi = 0; mi < 2; ++mi)
      for (int nj = 0; nj < 4; ++nj)
        for (int e = 0; e < 8; ++e) acc[mi][nj][e] = 0.f;

    stageB(nt, 0, 0);
    for (int t = 0; t < T; ++t) {
      const int cur = t & 1, nxt = cur ^ 1;
      if (t + 1 < T) stageB(nt, t + 1, nxt);
#if HAVE_TDM
      if (tid < 32) {
        if (t + 1 < T) TDM_WAIT(1);
        else           TDM_WAIT(0);
      }
#endif
      __syncthreads();

      v16bf af[2], bv[4];
#pragma unroll
      for (int mi = 0; mi < 2; ++mi)
        af[mi] = load_a_frag(
            rb + (long)(wm * 32 + mi * 16 + l15) * 512 + t * 32, lhalf);
#pragma unroll
      for (int nj = 0; nj < 4; ++nj)
        bv[nj] = *(const v16bf*)&bT[cur][wn * 64 + nj * 16 + l15][lhalf * 16];

#pragma unroll
      for (int mi = 0; mi < 2; ++mi)
#pragma unroll
        for (int nj = 0; nj < 4; ++nj)
          acc[mi][nj] = __builtin_amdgcn_wmma_f32_16x16x32_bf16(
              false, af[mi], false, bv[nj], (short)0, acc[mi][nj], false, false);

      __syncthreads();
    }

    // argmin(||c||^2 - 2 r.c): local min over 4 N-frags, ds_min_u64 reduce.
#pragma unroll
    for (int mi = 0; mi < 2; ++mi)
#pragma unroll
      for (int r = 0; r < 8; ++r) {
        int tl = wm * 32 + mi * 16 + lhalf * 8 + r;
        float bestv = 3.4e38f;
        int   besti = 0;
#pragma unroll
        for (int nj = 0; nj < 4; ++nj) {
          int code = nt * 128 + wn * 64 + nj * 16 + l15;
          float v = cbn[code] - 2.f * acc[mi][nj][r];
          if (v < bestv) { bestv = v; besti = code; }
        }
        unsigned long long key =
            ((unsigned long long)order_f32(bestv) << 32) | (unsigned)besti;
        atomicMin(&best[tl], key);
      }
  }
  __syncthreads();

  // gather selected code rows, update quantized & residual
  for (int e = tid; e < 128 * HDIM; e += 256) {
    int tl = e >> 9, k = e & (HDIM - 1);
    unsigned idx = (unsigned)(best[tl] & 0xFFFFFFFFu);
    long g = (tok0 + tl) * (long)HDIM + k;
    float c = cbf[(long)idx * HDIM + k];
    residual[g] = residual[g] - c;
    quant[g]    = first ? c : (quant[g] + c);
  }
  if (tid < 128)
    idx_out[tok0 + tid] = (float)(unsigned)(best[tid] & 0xFFFFFFFFu);
}

// ---------------------------------------------------------------------------
// Host orchestration (all launches on `stream`, graph-capture safe)
// ---------------------------------------------------------------------------
extern "C" void kernel_launch(void* const* d_in, const int* in_sizes, int n_in,
                              void* d_out, int out_size, void* d_ws, size_t ws_size,
                              hipStream_t stream) {
  (void)in_sizes; (void)n_in; (void)out_size; (void)ws_size;

  const float* x      = (const float*)d_in[0];
  const float* enc_w1 = (const float*)d_in[1];
  const float* enc_b1 = (const float*)d_in[2];
  const float* enc_w2 = (const float*)d_in[3];
  const float* enc_b2 = (const float*)d_in[4];
  const float* dec_w1 = (const float*)d_in[5];
  const float* dec_b1 = (const float*)d_in[6];
  const float* dec_w2 = (const float*)d_in[7];
  const float* dec_b2 = (const float*)d_in[8];
  const float* cbf    = (const float*)d_in[9];

  float* out   = (float*)d_out;
  float* quant = out;
  float* idxs  = out + (size_t)N_TOK * HDIM;
  float* recon = idxs + (size_t)NCB * N_TOK;

  char* ws = (char*)d_ws;
  size_t off = 0;
  float*  hid1 = (float*)(ws + off);  off += (size_t)N_TOK * 1024 * sizeof(float);
  float*  hres = (float*)(ws + off);  off += (size_t)N_TOK * HDIM * sizeof(float);
  __bf16* w1t  = (__bf16*)(ws + off); off += (size_t)1024 * D_IN * sizeof(__bf16);
  __bf16* w2t  = (__bf16*)(ws + off); off += (size_t)HDIM * 1024 * sizeof(__bf16);
  __bf16* d1t  = (__bf16*)(ws + off); off += (size_t)1024 * HDIM * sizeof(__bf16);
  __bf16* d2t  = (__bf16*)(ws + off); off += (size_t)D_IN * 1024 * sizeof(__bf16);
  __bf16* cbb  = (__bf16*)(ws + off); off += (size_t)NCB * KCODE * HDIM * sizeof(__bf16);
  float*  cbn  = (float*)(ws + off);  off += (size_t)NCB * KCODE * sizeof(float);

  dim3 blk(256);
  auto cvtT = [&](const float* s, __bf16* d, int Kd, int Nd) {
    long n = (long)Kd * Nd;
    cvt_transpose_kernel<<<(unsigned)((n + 255) / 256), blk, 0, stream>>>(s, d, Kd, Nd);
  };
  cvtT(enc_w1, w1t, D_IN, 1024);   // Wt[Nd][Kd]
  cvtT(enc_w2, w2t, 1024, HDIM);
  cvtT(dec_w1, d1t, HDIM, 1024);
  cvtT(dec_w2, d2t, 1024, D_IN);
  {
    int n = NCB * KCODE * HDIM;
    cvt_f32_bf16<<<(n + 255) / 256, blk, 0, stream>>>(cbf, cbb, n);
  }
  cb_norm_kernel<<<(NCB * KCODE + 255) / 256, blk, 0, stream>>>(cbf, cbn,
                                                                NCB * KCODE, HDIM);

  // encoder MLP
  gemm_bias_kernel<true ><<<dim3(1024 / 128, N_TOK / 128), blk, 0, stream>>>(
      x, w1t, enc_b1, hid1, D_IN, 1024);
  gemm_bias_kernel<false><<<dim3(HDIM / 128, N_TOK / 128), blk, 0, stream>>>(
      hid1, w2t, enc_b2, hres, 1024, HDIM);

  // RVQ stages (residual updated in place; 145 KB dynamic LDS per block,
  // legal on CDNA5's 320 KB-per-WGP LDS)
  constexpr unsigned RVQ_SMEM = 128 * 512 * 2 + 2 * 128 * 32 * 2 + 128 * 8;
  for (int s = 0; s < NCB; ++s)
    rvq_stage_kernel<<<N_TOK / 128, blk, RVQ_SMEM, stream>>>(
        hres, quant, idxs + (size_t)s * N_TOK,
        cbb + (size_t)s * KCODE * HDIM, cbn + (size_t)s * KCODE,
        cbf + (size_t)s * KCODE * HDIM, s == 0 ? 1 : 0);

  // decoder MLP
  gemm_bias_kernel<true ><<<dim3(1024 / 128, N_TOK / 128), blk, 0, stream>>>(
      quant, d1t, dec_b1, hid1, HDIM, 1024);
  gemm_bias_kernel<false><<<dim3(D_IN / 128, N_TOK / 128), blk, 0, stream>>>(
      hid1, d2t, dec_b2, recon, 1024, D_IN);
}
